// MockMambaBlock_87557203296774
// MI455X (gfx1250) — compile-verified, matrix operand order checked by reading, not run on hardware
//
#include <hip/hip_runtime.h>
#include <hip/hip_bf16.h>

// ---------------------------------------------------------------------------
// MockMambaBlock forward for MI455X (gfx1250, wave32, WMMA).
//   0) pack_b_bf16 x4 : weights -> bf16 WMMA-fragment order (K padded to 32)
//   1) gemm<4>: xz = x @ W_in                       (4096 x 4096, K=1024)
//   2) conv_silu: x_conv = silu(causal_dwconv(x_proj)+b)
//   3) gemm<2>: x_ssm = x_conv @ W_x                (4096 x 32,  K=2048)
//   4) gemm<4>: dt_lin = dt_in @ W_dt -> xz[:, :2048]  (K=16 padded to 32)
//   5) scan: selective scan, fused softplus, +D skip, *silu(z); in-place
//   6) gemm<4>: out = y_mid @ W_out                 (4096 x 1024, K=2048)
// GEMM inner loop is double-buffered: loads for k-block kb+1 are in flight
// while k-block kb's WMMAs execute (intra-wave latency hiding).
// ---------------------------------------------------------------------------

typedef __attribute__((ext_vector_type(16))) __bf16 v16bf;
typedef __attribute__((ext_vector_type(8)))  float  v8f;
typedef __attribute__((ext_vector_type(4)))  float  v4f;

#define BATCH   2
#define SEQLEN  2048
#define DMODEL  1024
#define DINNER  2048
#define DSTATE  16
#define DCONV   4
#define BL      (BATCH * SEQLEN)

// ---------------------------------------------------------------------------
// Pack f32 B[K,N] (row stride ldb) into bf16 fragments:
// Bp[((n0/16)*nkblk + k0/32)*512 + lane*16 + e] = B[k0 + kmap(e,g)][n0 + (l&15)]
// kmap(e,g) = e<8 ? 8g+e : 8+8g+e, g = lane>>4. k >= Ksrc packs 0.
// ---------------------------------------------------------------------------
__global__ __launch_bounds__(256)
void pack_b_bf16(const float* __restrict__ B, __bf16* __restrict__ Bp,
                 int N, int Ksrc, int Kpad, int ldb)
{
    const int t = blockIdx.x * blockDim.x + threadIdx.x;
    const int nkblk = Kpad >> 5;
    if (t >= (N >> 4) * nkblk * 32) return;
    const int lane  = t & 31;
    const int kblk  = (t >> 5) % nkblk;
    const int ntile = (t >> 5) / nkblk;
    const int g = lane >> 4;
    const int n = (ntile << 4) + (lane & 15);

    v16bf out;
    #pragma unroll
    for (int e = 0; e < 16; ++e) {
        const int k = (kblk << 5) + ((e < 8) ? (8 * g + e) : (8 + 8 * g + e));
        out[e] = (__bf16)((k < Ksrc) ? B[(size_t)k * ldb + n] : 0.0f);
    }
    *(v16bf*)(Bp + (size_t)t * 16) = out;
}

// ---------------------------------------------------------------------------
// Double-buffer helpers. Buffer holds one k-block: A raw f32 (converted at
// compute time so staging has no dependent waits) + NT packed B fragments.
// ---------------------------------------------------------------------------
template<int NT>
struct KBuf { v4f ra[4]; v16bf rb[NT]; };

template<int NT>
__device__ __forceinline__ void stage_kblk(const float* __restrict__ Arow,
                                           const __bf16* const (&Bb)[NT],
                                           int kb, KBuf<NT>& buf)
{
    const int ko = kb << 5;
    buf.ra[0] = *(const v4f*)(Arow + ko + 0);
    buf.ra[1] = *(const v4f*)(Arow + ko + 4);
    buf.ra[2] = *(const v4f*)(Arow + ko + 16);
    buf.ra[3] = *(const v4f*)(Arow + ko + 20);
    #pragma unroll
    for (int j = 0; j < NT; ++j)
        buf.rb[j] = *(const v16bf*)(Bb[j] + (size_t)kb * 512);
}

template<int NT>
__device__ __forceinline__ void mma_kblk(const KBuf<NT>& buf, v8f (&acc)[NT])
{
    v16bf a;
    #pragma unroll
    for (int e = 0; e < 4; ++e) {
        a[e]      = (__bf16)buf.ra[0][e];
        a[4 + e]  = (__bf16)buf.ra[1][e];
        a[8 + e]  = (__bf16)buf.ra[2][e];
        a[12 + e] = (__bf16)buf.ra[3][e];
    }
    #pragma unroll
    for (int j = 0; j < NT; ++j)
        acc[j] = __builtin_amdgcn_wmma_f32_16x16x32_bf16(
                     false, a, false, buf.rb[j], (short)0, acc[j], false, false);
}

// ---------------------------------------------------------------------------
// C[M,N] = A[M,K] * B[K,N]; A,C f32; B pre-packed bf16 fragments.
// One wave32 computes a 16 x (16*NT) strip; K multiple of 32 (pre-padded).
// No divergent control flow -> EXEC all-ones at every WMMA.
// ---------------------------------------------------------------------------
template<int NT>
__global__ __launch_bounds__(128)
void gemm_wmma_packed(const float* __restrict__ A, const __bf16* __restrict__ Bp,
                      float* __restrict__ C, int K, int lda, int ldc)
{
    const int lane  = threadIdx.x;          // 0..31
    const int g     = lane >> 4;
    const int mr    = lane & 15;
    const int tileM = (blockIdx.x * blockDim.y + threadIdx.y) * 16;
    const int tileN0 = blockIdx.y * (16 * NT);
    const int nkblk = K >> 5;

    const float* Arow = A + (size_t)(tileM + mr) * lda + 8 * g;
    const __bf16* Bb[NT];
    #pragma unroll
    for (int j = 0; j < NT; ++j)
        Bb[j] = Bp + ((size_t)((tileN0 >> 4) + j) * nkblk * 32 + lane) * 16;

    v8f acc[NT];
    #pragma unroll
    for (int j = 0; j < NT; ++j) acc[j] = (v8f){};

    KBuf<NT> buf0, buf1;
    stage_kblk<NT>(Arow, Bb, 0, buf0);

    int kb = 0;
    for (; kb + 2 <= nkblk; kb += 2) {
        __builtin_prefetch(Arow + ((kb + 3) << 5), 0, 3);   // speculative, 2 ahead
        stage_kblk<NT>(Arow, Bb, kb + 1, buf1);             // loads in flight ...
        mma_kblk<NT>(buf0, acc);                            // ... while computing
        if (kb + 2 < nkblk) stage_kblk<NT>(Arow, Bb, kb + 2, buf0);
        mma_kblk<NT>(buf1, acc);
    }
    if (kb < nkblk) mma_kblk<NT>(buf0, acc);                // odd tail (dt_proj)

    #pragma unroll
    for (int j = 0; j < NT; ++j)
        #pragma unroll
        for (int r = 0; r < 8; ++r)
            C[(size_t)(tileM + r + 8 * g) * ldc + tileN0 + 16 * j + mr] = acc[j][r];
}

// ---------------------------------------------------------------------------
// Causal depthwise conv (kernel 4) + bias + SiLU; reads x_proj half of xz.
// ---------------------------------------------------------------------------
__global__ __launch_bounds__(256)
void conv_silu_kernel(const float* __restrict__ xz, float* __restrict__ xconv,
                      const float* __restrict__ conv_w, const float* __restrict__ conv_b)
{
    const int idx = blockIdx.x * blockDim.x + threadIdx.x;
    if (idx >= BATCH * SEQLEN * DINNER) return;
    const int d = idx % DINNER;
    const int t = (idx / DINNER) % SEQLEN;
    const int b = idx / (DINNER * SEQLEN);

    const float* xp = xz + (size_t)b * SEQLEN * (2 * DINNER) + d;
    float acc = conv_b[d];
    #pragma unroll
    for (int k = 0; k < DCONV; ++k) {
        const int tt = t - (DCONV - 1) + k;      // y[t] = sum_k x[t-3+k]*w[k]
        const float v = (tt >= 0) ? xp[(size_t)tt * (2 * DINNER)] : 0.0f;
        acc += v * conv_w[d * DCONV + k];
    }
    xconv[idx] = acc / (1.0f + __expf(-acc));    // SiLU
}

// ---------------------------------------------------------------------------
// Selective scan: thread = (batch, channel, state). 16 states of one channel
// occupy 16 adjacent lanes; shfl_xor butterfly reduces over states each t.
// Fuses softplus(dt_lin + b_dt), the D-skip and the silu(z) gate.
// xcym read as x_conv[t][d], overwritten with y_mid[t][d] (read-before-write
// within the same wave -> in-place safe).
// ---------------------------------------------------------------------------
__global__ __launch_bounds__(256)
void scan_kernel(float* xcym,
                 const float* __restrict__ dtl,     // xz cols 0..Di-1, stride 2Di
                 const float* __restrict__ xssm,    // (B,L,32): cols 16..31 = B_ssm
                 const float* __restrict__ zsrc,    // xz cols Di..2Di-1, stride 2Di
                 const float* __restrict__ A_log,
                 const float* __restrict__ Dvec,
                 const float* __restrict__ b_dt)
{
    const int gid = blockIdx.x * blockDim.x + threadIdx.x;
    const int s   = gid & 15;
    const int d   = (gid >> 4) % DINNER;
    const int b   = gid / (16 * DINNER);

    const float Ads = -__expf(A_log[d * DSTATE + s]);
    const float Dd  = Dvec[d];
    const float bdt = b_dt[d];

    float*       pxc = xcym + (size_t)b * SEQLEN * DINNER + d;
    const float* pdt = dtl  + (size_t)b * SEQLEN * (2 * DINNER) + d;
    const float* pB  = xssm + (size_t)b * SEQLEN * 32 + DSTATE + s;
    const float* pz  = zsrc + (size_t)b * SEQLEN * (2 * DINNER) + d;

    float h = 0.0f;
    for (int t = 0; t < SEQLEN; ++t) {
        const float xv = pxc[(size_t)t * DINNER];
        const float dl = pdt[(size_t)t * (2 * DINNER)] + bdt;
        const float dt = (dl > 20.0f) ? dl : __logf(1.0f + __expf(dl));
        const float Bt = pB[(size_t)t * 32];
        h = h * __expf(dt * Ads) + dt * Bt * xv;

        float y = h;
        y += __shfl_xor(y, 1);
        y += __shfl_xor(y, 2);
        y += __shfl_xor(y, 4);
        y += __shfl_xor(y, 8);
        if (s == 0) {
            const float zv = pz[(size_t)t * (2 * DINNER)];
            pxc[(size_t)t * DINNER] =
                (y + xv * Dd) * (zv / (1.0f + __expf(-zv)));
        }
    }
}

// ---------------------------------------------------------------------------
extern "C" void kernel_launch(void* const* d_in, const int* in_sizes, int n_in,
                              void* d_out, int out_size, void* d_ws, size_t ws_size,
                              hipStream_t stream)
{
    const float* x      = (const float*)d_in[0];
    const float* W_in   = (const float*)d_in[1];
    const float* conv_w = (const float*)d_in[2];
    const float* conv_b = (const float*)d_in[3];
    const float* W_x    = (const float*)d_in[4];
    const float* W_dt   = (const float*)d_in[5];
    const float* b_dt   = (const float*)d_in[6];
    const float* A_log  = (const float*)d_in[7];
    const float* Dv     = (const float*)d_in[8];
    const float* W_out  = (const float*)d_in[9];
    float* out = (float*)d_out;

    // ---- workspace layout ----
    float* ws    = (float*)d_ws;
    float* xz    = ws;                                    // BL x 4096 f32
    float* xconv = xz + (size_t)BL * (2 * DINNER);        // BL x 2048 f32 (-> y_mid)
    float* xssm  = xconv + (size_t)BL * DINNER;           // BL x 32 f32
    __bf16* Wp_in  = (__bf16*)(xssm + (size_t)BL * 32);
    __bf16* Wp_x   = Wp_in  + (size_t)(2 * DINNER / 16) * (DMODEL / 32) * 512;
    __bf16* Wp_dt  = Wp_x   + (size_t)(2 * DSTATE / 16) * (DINNER / 32) * 512;
    __bf16* Wp_out = Wp_dt  + (size_t)(DINNER / 16) * 1 * 512;

    // ---- 0) pack weights to bf16 fragment order ----
    pack_b_bf16<<<(size_t)(2*DINNER/16) * (DMODEL/32) * 32 / 256, 256, 0, stream>>>(
        W_in, Wp_in, 2 * DINNER, DMODEL, DMODEL, 2 * DINNER);
    pack_b_bf16<<<(size_t)(2*DSTATE/16) * (DINNER/32) * 32 / 256, 256, 0, stream>>>(
        W_x, Wp_x, 2 * DSTATE, DINNER, DINNER, 2 * DSTATE);
    pack_b_bf16<<<(size_t)(DINNER/16) * 1 * 32 / 256, 256, 0, stream>>>(
        W_dt, Wp_dt, DINNER, DSTATE, 32, DINNER);        // K=16 padded to 32
    pack_b_bf16<<<(size_t)(DMODEL/16) * (DINNER/32) * 32 / 256, 256, 0, stream>>>(
        W_out, Wp_out, DMODEL, DINNER, DINNER, DMODEL);

    // ---- 1) xz = x @ W_in : M=4096 N=4096 K=1024 ----
    gemm_wmma_packed<4><<<dim3(BL / 64, (2 * DINNER) / 64), dim3(32, 4), 0, stream>>>(
        x, Wp_in, xz, DMODEL, DMODEL, 2 * DINNER);

    // ---- 2) x_conv = silu(dwconv(x_proj) + conv_b) ----
    conv_silu_kernel<<<(BATCH * SEQLEN * DINNER + 255) / 256, 256, 0, stream>>>(
        xz, xconv, conv_w, conv_b);

    // ---- 3) x_ssm = x_conv @ W_x : M=4096 N=32 K=2048 ----
    gemm_wmma_packed<2><<<dim3(BL / 64, 1), dim3(32, 4), 0, stream>>>(
        xconv, Wp_x, xssm, DINNER, DINNER, 2 * DSTATE);

    // ---- 4) dt_lin = dt_in @ W_dt : M=4096 N=2048, K=16 padded to 32.
    //      A = xssm (lda=32): k=16..31 reads B_ssm cols but they multiply
    //      packed zeros in Wp_dt, so the result is exact. C -> xz[:, :2048]. ----
    gemm_wmma_packed<4><<<dim3(BL / 64, DINNER / 64), dim3(32, 4), 0, stream>>>(
        xssm, Wp_dt, xz, 32, 2 * DSTATE, 2 * DINNER);

    // ---- 5) selective scan (y_mid overwrites xconv) ----
    scan_kernel<<<(BATCH * DINNER * DSTATE) / 256, 256, 0, stream>>>(
        xconv, xz, xssm, xz + DINNER, A_log, Dv, b_dt);

    // ---- 6) out = y_mid @ W_out : M=4096 N=1024 K=2048 ----
    gemm_wmma_packed<4><<<dim3(BL / 64, DMODEL / 64), dim3(32, 4), 0, stream>>>(
        xconv, Wp_out, out, DINNER, DINNER, DMODEL);
}
